// ChocolateNet_91156385890604
// MI455X (gfx1250) — matrix operand order repaired and checked by reading, not compiled, not run on hardware
//
#include <hip/hip_runtime.h>

typedef __attribute__((ext_vector_type(16))) _Float16 v16h;
typedef __attribute__((ext_vector_type(8)))  float    v8f;
typedef __attribute__((ext_vector_type(4)))  unsigned v4u;
typedef __attribute__((ext_vector_type(8)))  int      v8i;
typedef __attribute__((ext_vector_type(4)))  int      v4i;

#define LL 1024
#define DD 640
#define KK 32
#define MM 64
#define THRESH 0.2f

// ---- dynamic-LDS layout for k_main (float offsets; base of dynamic LDS == 0) ----
#define OFF_WE   0        // W_edge  34*64 = 2176
#define OFF_BE   2176     // b_edge  64
#define OFF_WOUT 2240     // w_out   64
#define OFF_MU   2304     // rbf_mu  32
#define OFF_SIG  2336     // rbf_sig 32
#define OFF_XYZ  2368     // xyz     1024*3 = 3072
#define OFF_P    5440     // P row   1024
#define OFF_RED  6464     // 8 waves * 3 partials (+pad)
#define SMEM_FLOATS 6496  // 25984 bytes

#if __has_builtin(__builtin_amdgcn_tensor_load_to_lds)
#define HAVE_TDM 1
// 1-D contiguous f32 copy global -> LDS via the Tensor Data Mover.
// D# group0: count=1, lds_addr, 57-bit global addr, type=2 ("image").
// D# group1: data_size=2 (4B), tensor_dim0=n, tensor_dim1=1, tile_dim0=n,
//            tile_dim1=1, tensor_dim0_stride=n. Groups 2/3 (and the trailing
// group) unused (zero). 6-arg builtin form on this toolchain.
__device__ __forceinline__ void tdm_copy_f32(unsigned lds_byte_off,
                                             const float* __restrict__ gsrc,
                                             int n) {
    unsigned long long ga = (unsigned long long)(const void*)gsrc;
    v4u g0 = { 1u,                                   // count = 1 valid descriptor
               lds_byte_off,                         // lds_addr [63:32]
               (unsigned)(ga & 0xFFFFFFFFull),       // global_addr [95:64]
               (unsigned)((ga >> 32) & 0x1FFFFFFull) // global_addr [120:96]
               | 0x80000000u };                      // type = 2 -> bits [127:126]
    v8i g1 = { 0x20000,                              // data_size = 2 (4 bytes)
               (int)((unsigned)(n & 0xFFFF) << 16),  // tensor_dim0 lo16 -> [31:16]
               (int)(((n >> 16) & 0xFFFF) | (1 << 16)), // dim0 hi16 | tensor_dim1=1
               (int)((unsigned)(n & 0xFFFF) << 16),  // dim1 hi=0 | tile_dim0 -> [127:112]
               1,                                    // tile_dim1 = 1, tile_dim2 = 0
               n,                                    // tensor_dim0_stride lo32
               0, 0 };
    v4i gz4 = { 0, 0, 0, 0 };
    v8i gz8 = { 0, 0, 0, 0, 0, 0, 0, 0 };
    __builtin_amdgcn_tensor_load_to_lds(g0, g1, gz4, gz4, gz8, 0);
}
#else
#define HAVE_TDM 0
#endif

// ---------------- kernel 1: node_proj = S @ W_node  (f16 WMMA) ------------------
// one wave per 16-row tile of S; stores lane-permuted layout:
// NP[row*64 + lr*4 + mt] = (S@Wn)[row][mt*16+lr]  -> b128-loadable per (row,lane).
__global__ void k_nodeproj(const float* __restrict__ S,
                           const float* __restrict__ Wn,
                           float* __restrict__ NP) {
    int wave = (blockIdx.x * blockDim.x + threadIdx.x) >> 5;
    int lane = threadIdx.x & 31;
    int i0 = wave * 16;
    int halfsel = lane >> 4;
    int lr = lane & 15;

    v8f c[4];
    #pragma unroll
    for (int mt = 0; mt < 4; ++mt) c[mt] = (v8f){0.f,0.f,0.f,0.f,0.f,0.f,0.f,0.f};

    for (int kc = 0; kc < DD; kc += 32) {
        v16h a;
        int kbaseA = halfsel * 8;
        const float* srow = S + (size_t)(i0 + lr) * DD + kc;
        #pragma unroll
        for (int t = 0; t < 16; ++t) {
            int kk = kbaseA + ((t < 8) ? t : (8 + t));
            a[t] = (_Float16)srow[kk];
        }
        int kbaseB = halfsel * 8;
        v16h b[4];
        #pragma unroll
        for (int t = 0; t < 16; ++t) {
            const float* wrow = Wn + (size_t)(kc + kbaseB + ((t < 8) ? t : (8 + t))) * MM;
            b[0][t] = (_Float16)wrow[ 0 + lr];
            b[1][t] = (_Float16)wrow[16 + lr];
            b[2][t] = (_Float16)wrow[32 + lr];
            b[3][t] = (_Float16)wrow[48 + lr];
        }
        #pragma unroll
        for (int mt = 0; mt < 4; ++mt)
            c[mt] = __builtin_amdgcn_wmma_f32_16x16x32_f16(
                        false, a, false, b[mt], (short)0, c[mt], false, false);
    }
    #pragma unroll
    for (int r = 0; r < 8; ++r) {
        int row = i0 + r + halfsel * 8;
        float* orow = NP + (size_t)row * MM + lr * 4;   // permuted: [row][lr][mt]
        #pragma unroll
        for (int mt = 0; mt < 4; ++mt) orow[mt] = c[mt][r];
    }
}

// ---------------- kernel 2: main edge-gated coordinate update -------------------
// one block per i (1024 blocks, 8 waves); wave w handles j-blocks w*8 .. w*8+7.
__global__ void __launch_bounds__(256)
k_main(const float* __restrict__ P,
       const float* __restrict__ xyz,
       const float* __restrict__ mu,
       const float* __restrict__ sig,
       const float* __restrict__ We,
       const float* __restrict__ be,
       const float* __restrict__ wout,
       const float* __restrict__ NP,
       float* __restrict__ out) {
    extern __shared__ float smem[];
    float* sWe   = smem + OFF_WE;
    float* sbe   = smem + OFF_BE;
    float* swout = smem + OFF_WOUT;
    float* smu   = smem + OFF_MU;
    float* ssig  = smem + OFF_SIG;
    float* sxyz  = smem + OFF_XYZ;
    float* sP    = smem + OFF_P;
    float* sred  = smem + OFF_RED;

    int tid = threadIdx.x;
    int i = blockIdx.x;

#if HAVE_TDM
    if (tid < 32) {
        tdm_copy_f32(OFF_WE  * 4, We, (KK + 2) * MM);
        tdm_copy_f32(OFF_XYZ * 4, xyz, LL * 3);
        tdm_copy_f32(OFF_P   * 4, P + (size_t)i * LL, LL);
        __builtin_amdgcn_s_wait_tensorcnt(0);
    }
#else
    for (int t = tid; t < (KK + 2) * MM; t += 256) sWe[t] = We[t];
    for (int t = tid; t < LL * 3; t += 256)       sxyz[t] = xyz[t];
    for (int t = tid; t < LL; t += 256)           sP[t] = P[(size_t)i * LL + t];
#endif
    for (int t = tid; t < MM; t += 256) { sbe[t] = be[t]; swout[t] = wout[t]; }
    for (int t = tid; t < KK; t += 256) { smu[t] = mu[t]; ssig[t] = sig[t]; }
    __syncthreads();

    int wv = tid >> 5;
    int lane = tid & 31;
    int halfsel = lane >> 4;
    int lr = lane & 15;

    float xi0 = sxyz[i * 3 + 0], xi1 = sxyz[i * 3 + 1], xi2 = sxyz[i * 3 + 2];

    // per-lane RBF constants (A-fragment K slice per ISA layout)
    float muv[16], nis[16];
    int kbaseA = halfsel * 8;
    #pragma unroll
    for (int t = 0; t < 16; ++t) {
        int kk = kbaseA + ((t < 8) ? t : (8 + t));
        muv[t] = smu[kk];
        float sg = ssig[kk];
        nis[t] = -1.0f / (2.0f * sg * sg);
    }

    // B fragments (persist across all 8 j-blocks of this wave)
    int kbaseB = halfsel * 8;
    v16h b[4];
    #pragma unroll
    for (int t = 0; t < 16; ++t) {
        const float* wr = sWe + (2 + kbaseB + ((t < 8) ? t : (8 + t))) * MM;
        b[0][t] = (_Float16)wr[ 0 + lr];
        b[1][t] = (_Float16)wr[16 + lr];
        b[2][t] = (_Float16)wr[32 + lr];
        b[3][t] = (_Float16)wr[48 + lr];
    }
    // per-lane column constants
    float w0m[4], w1m[4], bem[4], wom[4];
    #pragma unroll
    for (int mt = 0; mt < 4; ++mt) {
        int m = mt * 16 + lr;
        w0m[mt] = sWe[m];
        w1m[mt] = sWe[MM + m];
        bem[mt] = sbe[m];
        wom[mt] = swout[m];
    }

    float px = 0.f, py = 0.f, pz = 0.f;

    for (int it = 0; it < 8; ++it) {
        int jb = wv * 8 + it;
        int j0 = jb << 4;

        // ---- A fragment: RBF features of row j = j0 + lr
        int jA = j0 + lr;
        float r0 = sxyz[jA * 3 + 0] - xi0;
        float r1 = sxyz[jA * 3 + 1] - xi1;
        float r2 = sxyz[jA * 3 + 2] - xi2;
        float dA = sqrtf(r0 * r0 + r1 * r1 + r2 * r2 + 1e-12f);
        v16h a;
        #pragma unroll
        for (int t = 0; t < 16; ++t) {
            float dm = dA - muv[t];
            a[t] = (_Float16)__expf(dm * dm * nis[t]);
        }

        // ---- per-row edge scalars (rows r=0..7 -> j = j0 + r + halfsel*8)
        float etypev[8], pijv[8], maskv[8];
        #pragma unroll
        for (int r = 0; r < 8; ++r) {
            int j = j0 + r + halfsel * 8;
            float p = sP[j];
            bool bb = (j == i + 1);
            bool ct = (p > THRESH) && (j > i + 2);
            etypev[r] = ct ? 1.0f : 0.0f;
            pijv[r]   = bb ? 1.0f : p;
            maskv[r]  = (bb || ct) ? 1.0f : 0.0f;
        }

        // ---- C init: etype*We[0,m] + pij*We[1,m] + b[m] + node_proj[j,m]
        v8f c[4];
        #pragma unroll
        for (int r = 0; r < 8; ++r) {
            int j = j0 + r + halfsel * 8;
            float4 np = ((const float4*)NP)[j * 16 + lr];   // permuted layout, b128
            const float* npv = (const float*)&np;
            #pragma unroll
            for (int mt = 0; mt < 4; ++mt)
                c[mt][r] = etypev[r] * w0m[mt] + pijv[r] * w1m[mt] + bem[mt] + npv[mt];
        }

        // ---- WMMA: rank-32 RBF contraction into the 4 msg tiles
        #pragma unroll
        for (int mt = 0; mt < 4; ++mt)
            c[mt] = __builtin_amdgcn_wmma_f32_16x16x32_f16(
                        false, a, false, b[mt], (short)0, c[mt], false, false);

        // ---- relu, dot w_out, cross-lane reduce, sigmoid*mask, gated vec sum
        #pragma unroll
        for (int r = 0; r < 8; ++r) {
            float s = fmaxf(c[0][r], 0.f) * wom[0] + fmaxf(c[1][r], 0.f) * wom[1]
                    + fmaxf(c[2][r], 0.f) * wom[2] + fmaxf(c[3][r], 0.f) * wom[3];
            s += __shfl_xor(s, 1);
            s += __shfl_xor(s, 2);
            s += __shfl_xor(s, 4);
            s += __shfl_xor(s, 8);      // full row sum within each 16-lane half
            float g = maskv[r] / (1.0f + __expf(-s));
            int j = j0 + r + halfsel * 8;
            px += g * (sxyz[j * 3 + 0] - xi0);
            py += g * (sxyz[j * 3 + 1] - xi1);
            pz += g * (sxyz[j * 3 + 2] - xi2);
        }
    }

    // fold the two lane-halves, one partial per wave
    px += __shfl_xor(px, 16);
    py += __shfl_xor(py, 16);
    pz += __shfl_xor(pz, 16);
    if (lane == 0) {
        sred[wv * 3 + 0] = px;
        sred[wv * 3 + 1] = py;
        sred[wv * 3 + 2] = pz;
    }
    __syncthreads();
    if (tid < 3) {
        float acc = sxyz[i * 3 + tid];
        #pragma unroll
        for (int w = 0; w < 8; ++w) acc += sred[w * 3 + tid];
        out[i * 3 + tid] = acc;
    }
}

extern "C" void kernel_launch(void* const* d_in, const int* in_sizes, int n_in,
                              void* d_out, int out_size, void* d_ws, size_t ws_size,
                              hipStream_t stream) {
    (void)in_sizes; (void)n_in; (void)out_size; (void)ws_size;
    const float* S    = (const float*)d_in[0];   // (1024,640)
    const float* P    = (const float*)d_in[1];   // (1024,1024)
    const float* xyz  = (const float*)d_in[2];   // (1024,3)
    const float* mu   = (const float*)d_in[3];   // (32,)
    const float* sig  = (const float*)d_in[4];   // (32,)
    const float* We   = (const float*)d_in[5];   // (34,64)
    const float* be   = (const float*)d_in[6];   // (64,)
    const float* Wn   = (const float*)d_in[7];   // (640,64)
    const float* wout = (const float*)d_in[8];   // (64,)
    float* out = (float*)d_out;                  // (1024,3)
    float* NP  = (float*)d_ws;                   // node_proj (1024,64) f32, permuted

    // 1) node_proj: 64 row-tiles -> 64 waves -> 8 blocks x 256 threads
    k_nodeproj<<<8, 256, 0, stream>>>(S, Wn, NP);

    // 2) main: one block per i, 8 waves, 32 WMMAs/wave, TDM-staged LDS
    k_main<<<LL, 256, SMEM_FLOATS * sizeof(float), stream>>>(
        P, xyz, mu, sig, We, be, wout, NP, out);
}
